// MultiHead_85667417686380
// MI455X (gfx1250) — matrix operand order repaired and checked by reading, not compiled, not run on hardware
//
#include <hip/hip_runtime.h>
#include <hip/hip_bf16.h>
#include <stdint.h>

#define BB 4
#define TT 1024
#define EE 1024
#define HH 16
#define HS 64

typedef __attribute__((ext_vector_type(16))) __bf16 v16bf;
typedef __attribute__((ext_vector_type(8)))  float  v8f;
typedef unsigned short ushort_t;

__device__ __forceinline__ ushort_t f2bf(float f) {
  unsigned int u = __float_as_uint(f);
  unsigned int r = u + 0x7FFFu + ((u >> 16) & 1u);  // round-to-nearest-even
  return (ushort_t)(r >> 16);
}

union FragBF {
  uint4  q[2];
  v16bf  v;
};

// A-fragment (16x32 bf16, MxK): lane(half,col) holds row=col,
// K = half*8+[0..7] (v0..v3) and 16+half*8+[0..7] (v4..v7).
// p = row_ptr + k_chunk_base
__device__ __forceinline__ v16bf load_a_frag(const ushort_t* p, int half) {
  FragBF f;
  f.q[0] = *(const uint4*)(p + half * 8);
  f.q[1] = *(const uint4*)(p + 16 + half * 8);
  return f.v;
}

// B-fragment (32x16 bf16, KxN), loaded column-major (= row of B^T):
// lane(half,col) holds column=col, K = half*8+[0..7] and 16+half*8+[0..7].
// p = colrow_ptr (row of B^T) + k_chunk_base
__device__ __forceinline__ v16bf load_b_frag(const ushort_t* p, int half) {
  FragBF f;
  const ushort_t* p2 = p + half * 8;
  f.q[0] = *(const uint4*)(p2);
  f.q[1] = *(const uint4*)(p2 + 16);
  return f.v;
}

// ---------------- LayerNorm (fp32 -> bf16) ----------------
__global__ __launch_bounds__(256)
void ln_bf16_kernel(const float* __restrict__ x, const float* __restrict__ g,
                    const float* __restrict__ bta, ushort_t* __restrict__ out) {
  int row = blockIdx.x;
  int tid = threadIdx.x;
  const float* xr = x + (size_t)row * EE;
  __shared__ float red[256];
  float v[4];
  float s = 0.f;
#pragma unroll
  for (int i = 0; i < 4; ++i) { v[i] = xr[tid + i * 256]; s += v[i]; }
  red[tid] = s; __syncthreads();
  for (int o = 128; o > 0; o >>= 1) { if (tid < o) red[tid] += red[tid + o]; __syncthreads(); }
  float mu = red[0] * (1.0f / EE);
  __syncthreads();
  float s2 = 0.f;
#pragma unroll
  for (int i = 0; i < 4; ++i) { float d = v[i] - mu; s2 += d * d; }
  red[tid] = s2; __syncthreads();
  for (int o = 128; o > 0; o >>= 1) { if (tid < o) red[tid] += red[tid + o]; __syncthreads(); }
  float rstd = rsqrtf(red[0] * (1.0f / EE) + 1e-5f);
#pragma unroll
  for (int i = 0; i < 4; ++i) {
    int c = tid + i * 256;
    out[(size_t)row * EE + c] = f2bf((v[i] - mu) * rstd * g[c] + bta[c]);
  }
}

// ---------------- Weight transpose + bf16 convert: WT[n][k] = W[k][n] ----------------
__global__ __launch_bounds__(256)
void transpose_bf16_kernel(const float* __restrict__ W, ushort_t* __restrict__ WT) {
  int idx = blockIdx.x * 256 + threadIdx.x;  // over EE*EE
  int n = idx >> 10;
  int k = idx & 1023;
  WT[idx] = f2bf(W[(size_t)k * EE + n]);
}

// ---------------- WMMA GEMM: C[4096 x 1024] = A[4096 x 1024] * B, B given as B^T ----
// One wave computes a 32(M) x 64(N) tile: each B fragment is reused by two A
// fragments (8 WMMA per 12 uint4 loads). MODE 0: bf16 out [B,H,T,HS];
// MODE 1: bf16 out [B,H,HS,T]; MODE 2: fp32 out [M,N] + bias.
template <int MODE>
__global__ __launch_bounds__(32)
void gemm_wmma_kernel(const ushort_t* __restrict__ A, const ushort_t* __restrict__ BT,
                      const float* __restrict__ bias, void* __restrict__ outp) {
  int n0 = blockIdx.x * 64;
  int m0 = blockIdx.y * 32;
  int lane = threadIdx.x;
  int col = lane & 15, half = lane >> 4;
  v8f acc[2][4] = {};
  const ushort_t* aRow0 = A + (size_t)(m0 + col) * EE;
  const ushort_t* aRow1 = aRow0 + (size_t)16 * EE;
  for (int kk = 0; kk < EE; kk += 32) {
    v16bf a0 = load_a_frag(aRow0 + kk, half);
    v16bf a1 = load_a_frag(aRow1 + kk, half);
#pragma unroll
    for (int f = 0; f < 4; ++f) {
      v16bf bb = load_b_frag(BT + (size_t)(n0 + f * 16 + col) * EE + kk, half);
      acc[0][f] = __builtin_amdgcn_wmma_f32_16x16x32_bf16(false, a0, false, bb,
                                                          (short)0, acc[0][f], false, false);
      acc[1][f] = __builtin_amdgcn_wmma_f32_16x16x32_bf16(false, a1, false, bb,
                                                          (short)0, acc[1][f], false, false);
    }
  }
#pragma unroll
  for (int mi = 0; mi < 2; ++mi) {
#pragma unroll
    for (int f = 0; f < 4; ++f) {
      int n = n0 + f * 16 + col;
#pragma unroll
      for (int r = 0; r < 8; ++r) {
        int m = m0 + mi * 16 + r + 8 * half;  // C layout: row = r + 8*half
        float val = acc[mi][f][r];
        if constexpr (MODE == 2) {
          ((float*)outp)[(size_t)m * EE + n] = val + bias[n];
        } else {
          int b = m >> 10, t = m & 1023;
          int h = n >> 6, d = n & 63;
          ushort_t* ob = (ushort_t*)outp;
          if constexpr (MODE == 0)
            ob[(((size_t)(b * HH + h) * TT + t) * HS + d)] = f2bf(val);  // [B,H,T,HS]
          else
            ob[(((size_t)(b * HH + h) * HS + d) * TT + t)] = f2bf(val);  // [B,H,HS,T]
        }
      }
    }
  }
}

// ---------------- Causal flash attention: one wave per (b,h,16-query block) --------
__global__ __launch_bounds__(32)
void attn_kernel(const ushort_t* __restrict__ qh, const ushort_t* __restrict__ kh,
                 const ushort_t* __restrict__ vhT, ushort_t* __restrict__ attnC) {
  int bi = blockIdx.x;           // B*H*(T/16) = 4096
  int qblk = bi & 63;
  int h = (bi >> 6) & 15;
  int b = bi >> 10;
  int qb = qblk * 16;
  int lane = threadIdx.x;
  int col = lane & 15, half = lane >> 4;

  const ushort_t* qB = qh + (size_t)(b * HH + h) * TT * HS;   // [T,HS]
  const ushort_t* kB = kh + (size_t)(b * HH + h) * TT * HS;   // [T,HS]
  const ushort_t* vB = vhT + (size_t)(b * HH + h) * HS * TT;  // [HS,T]

  // Q tile A-fragments, K-dim HS=64 in two chunks of 32
  v16bf aq0 = load_a_frag(qB + (size_t)(qb + col) * HS + 0, half);
  v16bf aq1 = load_a_frag(qB + (size_t)(qb + col) * HS + 32, half);

  v8f acc[4] = {};
  float mrun[8], lrun[8];
#pragma unroll
  for (int r = 0; r < 8; ++r) { mrun[r] = -1e30f; lrun[r] = 0.f; }

  __shared__ __align__(16) ushort_t pT[16 * 32];  // P tile bounce, bf16

  const float scale = 0.03125f;  // 1/sqrt(E) = 1/32
  int nkt = (qb + 16 + 31) >> 5; // 32-key tiles up to causal limit
  for (int t32 = 0; t32 < nkt; ++t32) {
    int k0 = t32 << 5;
    v8f s[2];
#pragma unroll
    for (int fn = 0; fn < 2; ++fn) {
      v16bf bk0 = load_b_frag(kB + (size_t)(k0 + fn * 16 + col) * HS + 0, half);
      v16bf bk1 = load_b_frag(kB + (size_t)(k0 + fn * 16 + col) * HS + 32, half);
      v8f c = {};
      c = __builtin_amdgcn_wmma_f32_16x16x32_bf16(false, aq0, false, bk0, (short)0, c, false, false);
      c = __builtin_amdgcn_wmma_f32_16x16x32_bf16(false, aq1, false, bk1, (short)0, c, false, false);
      s[fn] = c;
    }
    float alpha[8];
#pragma unroll
    for (int r = 0; r < 8; ++r) {
      int qi = qb + r + 8 * half;
      float v0 = s[0][r] * scale;
      float v1 = s[1][r] * scale;
      if (k0 + col > qi)      v0 = -1e30f;   // causal mask
      if (k0 + 16 + col > qi) v1 = -1e30f;
      float mx = fmaxf(v0, v1);
      mx = fmaxf(mx, __shfl_xor(mx, 1, 32));
      mx = fmaxf(mx, __shfl_xor(mx, 2, 32));
      mx = fmaxf(mx, __shfl_xor(mx, 4, 32));
      mx = fmaxf(mx, __shfl_xor(mx, 8, 32));
      float mnew = fmaxf(mrun[r], mx);
      float al = __expf(mrun[r] - mnew);
      float e0 = __expf(v0 - mnew);
      float e1 = __expf(v1 - mnew);
      float ts = e0 + e1;
      ts += __shfl_xor(ts, 1, 32);
      ts += __shfl_xor(ts, 2, 32);
      ts += __shfl_xor(ts, 4, 32);
      ts += __shfl_xor(ts, 8, 32);
      lrun[r] = lrun[r] * al + ts;
      mrun[r] = mnew;
      alpha[r] = al;
      int prow = r + 8 * half;
      pT[prow * 32 + col]      = f2bf(e0);
      pT[prow * 32 + 16 + col] = f2bf(e1);
    }
#pragma unroll
    for (int f = 0; f < 4; ++f)
#pragma unroll
      for (int r = 0; r < 8; ++r) acc[f][r] *= alpha[r];
    __syncthreads();
    // P (16x32) as A-fragment from LDS; V columns from vhT are contiguous in T
    v16bf pa = load_a_frag(pT + col * 32, half);
#pragma unroll
    for (int f = 0; f < 4; ++f) {
      v16bf bv = load_b_frag(vB + (size_t)(f * 16 + col) * TT + k0, half);
      acc[f] = __builtin_amdgcn_wmma_f32_16x16x32_bf16(false, pa, false, bv,
                                                       (short)0, acc[f], false, false);
    }
    __syncthreads();
  }
#pragma unroll
  for (int r = 0; r < 8; ++r) {
    float inv = 1.0f / lrun[r];
    int t = qb + r + 8 * half;
#pragma unroll
    for (int f = 0; f < 4; ++f) {
      attnC[((size_t)b * TT + t) * EE + h * HS + f * 16 + col] = f2bf(acc[f][r] * inv);
    }
  }
}

extern "C" void kernel_launch(void* const* d_in, const int* in_sizes, int n_in,
                              void* d_out, int out_size, void* d_ws, size_t ws_size,
                              hipStream_t stream) {
  (void)in_sizes; (void)n_in; (void)out_size; (void)ws_size;
  const float* v    = (const float*)d_in[0];
  const float* k    = (const float*)d_in[1];
  const float* q    = (const float*)d_in[2];
  const float* ln_g = (const float*)d_in[3];
  const float* ln_b = (const float*)d_in[4];
  const float* Wq   = (const float*)d_in[5];
  const float* Wk   = (const float*)d_in[6];
  const float* Wv   = (const float*)d_in[7];
  const float* Wp   = (const float*)d_in[8];
  const float* bp   = (const float*)d_in[9];

  const size_t ROWS = (size_t)BB * TT;   // 4096
  const size_t XN   = ROWS * EE;         // 4M elems
  const size_t WSZ  = (size_t)EE * EE;   // 1M elems
  ushort_t* wsp  = (ushort_t*)d_ws;
  ushort_t* qn    = wsp;
  ushort_t* kn    = qn + XN;
  ushort_t* vn    = kn + XN;
  ushort_t* WqT   = vn + XN;
  ushort_t* WkT   = WqT + WSZ;
  ushort_t* WvT   = WkT + WSZ;
  ushort_t* WpT   = WvT + WSZ;
  ushort_t* qhB   = WpT + WSZ;   // [B,H,T,HS]
  ushort_t* khB   = qhB + XN;    // [B,H,T,HS]
  ushort_t* vhTb  = khB + XN;    // [B,H,HS,T]
  ushort_t* attnC = vhTb + XN;   // [B,T,E] (heads concat)

  ln_bf16_kernel<<<(int)ROWS, 256, 0, stream>>>(q, ln_g, ln_b, qn);
  ln_bf16_kernel<<<(int)ROWS, 256, 0, stream>>>(k, ln_g, ln_b, kn);
  ln_bf16_kernel<<<(int)ROWS, 256, 0, stream>>>(v, ln_g, ln_b, vn);

  transpose_bf16_kernel<<<4096, 256, 0, stream>>>(Wq, WqT);
  transpose_bf16_kernel<<<4096, 256, 0, stream>>>(Wk, WkT);
  transpose_bf16_kernel<<<4096, 256, 0, stream>>>(Wv, WvT);
  transpose_bf16_kernel<<<4096, 256, 0, stream>>>(Wp, WpT);

  dim3 g(EE / 64, (unsigned)(ROWS / 32));
  gemm_wmma_kernel<0><<<g, 32, 0, stream>>>(qn, WqT, nullptr, qhB);
  gemm_wmma_kernel<0><<<g, 32, 0, stream>>>(kn, WkT, nullptr, khB);
  gemm_wmma_kernel<1><<<g, 32, 0, stream>>>(vn, WvT, nullptr, vhTb);

  attn_kernel<<<BB * HH * (TT / 16), 32, 0, stream>>>(qhB, khB, vhTb, attnC);

  gemm_wmma_kernel<2><<<g, 32, 0, stream>>>(attnC, WpT, bp, d_out);
}
